// SoftTemporalAttention_22789096473267
// MI455X (gfx1250) — matrix operand order repaired and checked by reading, not compiled, not run on hardware
//
#include <hip/hip_runtime.h>
#include <hip/hip_bf16.h>

// ---------------- CDNA5 WMMA types ----------------
typedef __attribute__((ext_vector_type(16))) _Float16 v16h;
typedef __attribute__((ext_vector_type(8)))  _Float16 v8h;
typedef __attribute__((ext_vector_type(8)))  float    v8f;

union HFrag { v16h v; v8h h[2]; };

// A fragment (16x32 f16, M x K), row-major source with stride `stride` halfs.
// ISA 7.12.2: lanes 0-15 -> M=lane, chunks K=[0..7],[16..23]; lanes 16-31 -> K=[8..15],[24..31].
static __device__ __forceinline__ v16h ld_a(const _Float16* base, int stride, int lane, int kk) {
  int row = lane & 15;
  int kb  = (lane >> 4) ? 8 : 0;
  const _Float16* p = base + row * stride + kk + kb;
  HFrag u;
  u.h[0] = *(const v8h*)(p);
  u.h[1] = *(const v8h*)(p + 16);
  return u.v;
}

// B fragment (32x16 f16, K x N) from TRANSPOSED weights [N][K] with stride `stride` halfs.
// Lane n<16 holds B[K=0..7,16..23][N=n]; lane>=16 holds B[K=8..15,24..31][N=lane-16] (same K order as A).
static __device__ __forceinline__ v16h ld_b(const _Float16* baseT, int stride, int lane, int kk) {
  int col = lane & 15;
  int kb  = (lane >> 4) ? 8 : 0;
  const _Float16* p = baseT + col * stride + kk + kb;
  HFrag u;
  u.h[0] = *(const v8h*)(p);
  u.h[1] = *(const v8h*)(p + 16);
  return u.v;
}

// ---------------- Kernel 0: weights f32 -> f16 transposed in workspace ----------------
// W1 [384][256] -> W1T [256][384]; W2 [256][128] -> W2T [128][256]; Wa1 [128][64] -> Wa1T [64][128]
__global__ void convert_w(const float* __restrict__ W1, const float* __restrict__ W2,
                          const float* __restrict__ Wa1,
                          _Float16* __restrict__ W1T, _Float16* __restrict__ W2T,
                          _Float16* __restrict__ Wa1T) {
  int idx = blockIdx.x * 256 + threadIdx.x;
  if (idx < 98304) {
    int col = idx / 384, k = idx % 384;
    W1T[idx] = (_Float16)W1[k * 256 + col];
  } else if (idx < 131072) {
    int e = idx - 98304;
    int col = e / 256, k = e % 256;
    W2T[e] = (_Float16)W2[k * 128 + col];
  } else if (idx < 139264) {
    int e = idx - 131072;
    int col = e / 128, k = e % 128;
    Wa1T[e] = (_Float16)Wa1[k * 64 + col];
  }
}

// ---------------- Kernel 1: fused MLP chain, 64 rows per block ----------------
// Stage1: h=GELU(X@W1+b1)   (WMMA, K=384)
// Stage2: f=h@W2+b2         (WMMA, K=256)
// Stage3: fused=LayerNorm(f)*g+b  -> global fused[B*T][128] + f16 copy in LDS
// Stage4: a=tanh(fused@Wa1+ba1)   (WMMA, K=128)
// Stage5: logits = a.Wa2 + ba2    (VALU dot)
__global__ __launch_bounds__(256)
void fuse_kernel(const float* __restrict__ X,
                 const _Float16* __restrict__ W1T,
                 const _Float16* __restrict__ W2T,
                 const _Float16* __restrict__ Wa1T,
                 const float* __restrict__ b1, const float* __restrict__ b2,
                 const float* __restrict__ gamma, const float* __restrict__ beta,
                 const float* __restrict__ ba1,
                 const float* __restrict__ Wa2, const float* __restrict__ ba2,
                 float* __restrict__ fused_out, float* __restrict__ logits_out) {
  __shared__ __align__(16) char smem[65536];
  // Region reuse plan (all hand-synced with __syncthreads):
  _Float16* sH  = (_Float16*)(smem);          // [64][256] f16  h        (stage1 out / stage2 A)
  float*    sF  = (float*)(smem);             // [64][128] f32  f        (stage2 out, over sH)
  float*    sA  = (float*)(smem);             // [64][64]  f32  tanh(a)  (stage4 out, over sF)
  _Float16* sX  = (_Float16*)(smem + 32768);  // [64][32]  f16  X panel
  _Float16* sW1 = (_Float16*)(smem + 36864);  // [256][32] f16  W1T panel
  _Float16* sW2 = (_Float16*)(smem + 32768);  // [128][32] f16  W2T panel (over sX)
  _Float16* sFu = (_Float16*)(smem + 32768);  // [64][128] f16  fused    (over sW2)
  _Float16* sWa = (_Float16*)(smem + 49152);  // [64][128] f16  Wa1T     (over sW1 tail)

  const int tid   = threadIdx.x;
  const int lane  = tid & 31;
  const int wave  = tid >> 5;      // 8 waves
  const int rtile = wave >> 1;     // 0..3 : 16-row tile
  const int chalf = wave & 1;      // 0..1 : N half
  const long row0 = (long)blockIdx.x * 64;

  // ---------------- Stage 1: GEMM1 + GELU ----------------
  v8f acc1[8];
#pragma unroll
  for (int n = 0; n < 8; n++) acc1[n] = {};
  for (int kt = 0; kt < 12; kt++) {
    const int kk = kt * 32;
    {  // X panel [64][32]: f32 -> f16, coalesced (8 consecutive f32 per thread)
      int e = tid * 8, r = e >> 5, j = e & 31;
      const float* src = X + (row0 + r) * 384 + kk + j;
      _Float16* dst = sX + r * 32 + j;
#pragma unroll
      for (int i = 0; i < 8; i++) dst[i] = (_Float16)src[i];
      if (kt + 1 < 12) __builtin_prefetch(src + 32, 0, 1);  // global_prefetch_b8 next panel
    }
    {  // W1T panel [256 cols][32 k]: one 64B col-row per thread
      const _Float16* src = W1T + tid * 384 + kk;
      v8h* dst = (v8h*)(sW1 + tid * 32);
#pragma unroll
      for (int i = 0; i < 4; i++) dst[i] = *(const v8h*)(src + i * 8);
    }
    __syncthreads();
    v16h a = ld_a(sX + rtile * 16 * 32, 32, lane, 0);
#pragma unroll
    for (int n = 0; n < 8; n++) {
      v16h b = ld_b(sW1 + (chalf * 8 + n) * 16 * 32, 32, lane, 0);
      acc1[n] = __builtin_amdgcn_wmma_f32_16x16x32_f16(false, a, false, b,
                                                       (short)0, acc1[n], false, false);
    }
    __syncthreads();
  }
  {  // epilogue: + b1, exact GELU, store h as f16 (C/D layout: M = v + 8*(lane>=16))
    int rbase = rtile * 16 + ((lane >> 4) ? 8 : 0);
    int cbase = chalf * 128 + (lane & 15);
#pragma unroll
    for (int n = 0; n < 8; n++) {
      int col = cbase + n * 16;
      float bb = b1[col];
#pragma unroll
      for (int v = 0; v < 8; v++) {
        float x = acc1[n][v] + bb;
        float g = 0.5f * x * (1.0f + erff(x * 0.70710678118f));
        sH[(rbase + v) * 256 + col] = (_Float16)g;
      }
    }
  }
  __syncthreads();

  // ---------------- Stage 2: GEMM2 ----------------
  v8f acc2[4];
#pragma unroll
  for (int n = 0; n < 4; n++) acc2[n] = {};
  for (int kt = 0; kt < 8; kt++) {
    const int kk = kt * 32;
    {  // W2T panel [128][32]
      int col = tid >> 1, j0 = (tid & 1) * 16;
      const v8h* src = (const v8h*)(W2T + col * 256 + kk + j0);
      v8h* dst = (v8h*)(sW2 + col * 32 + j0);
      dst[0] = src[0];
      dst[1] = src[1];
    }
    __syncthreads();
    v16h a = ld_a(sH + rtile * 16 * 256, 256, lane, kk);
#pragma unroll
    for (int n = 0; n < 4; n++) {
      v16h b = ld_b(sW2 + (chalf * 4 + n) * 16 * 32, 32, lane, 0);
      acc2[n] = __builtin_amdgcn_wmma_f32_16x16x32_f16(false, a, false, b,
                                                       (short)0, acc2[n], false, false);
    }
    __syncthreads();
  }
  {  // epilogue: + b2, store f (f32) over h region (all h reads done at last barrier)
    int rbase = rtile * 16 + ((lane >> 4) ? 8 : 0);
    int cbase = chalf * 64 + (lane & 15);
#pragma unroll
    for (int n = 0; n < 4; n++) {
      int col = cbase + n * 16;
      float bb = b2[col];
#pragma unroll
      for (int v = 0; v < 8; v++) sF[(rbase + v) * 128 + col] = acc2[n][v] + bb;
    }
  }
  __syncthreads();

  // ---------------- Stage 3: LayerNorm (4 lanes per row, shfl reduce) ----------------
  {
    int r = tid >> 2, q = tid & 3;
    const float* frow = sF + r * 128 + q * 32;
    float s = 0.f, s2 = 0.f;
#pragma unroll
    for (int i = 0; i < 32; i++) { float x = frow[i]; s += x; s2 += x * x; }
    s += __shfl_xor(s, 1, 32);  s2 += __shfl_xor(s2, 1, 32);
    s += __shfl_xor(s, 2, 32);  s2 += __shfl_xor(s2, 2, 32);
    float mu  = s * (1.0f / 128.0f);
    float var = s2 * (1.0f / 128.0f) - mu * mu;
    float inv = rsqrtf(var + 1e-5f);
    float* gout = fused_out + (row0 + r) * 128 + q * 32;
#pragma unroll
    for (int i = 0; i < 32; i++) {
      int d = q * 32 + i;
      float y = (frow[i] - mu) * inv * gamma[d] + beta[d];
      sFu[r * 128 + d] = (_Float16)y;
      gout[i] = y;  // kernel 2 gathers top-32 rows from here
    }
  }
  {  // stage in Wa1T [64][128] f16 (stays for the whole K loop)
    int e = tid * 32, col = e >> 7, j = e & 127;
    const v8h* src = (const v8h*)(Wa1T + col * 128 + j);
    v8h* dst = (v8h*)(sWa + col * 128 + j);
#pragma unroll
    for (int i = 0; i < 4; i++) dst[i] = src[i];
  }
  __syncthreads();

  // ---------------- Stage 4: GEMM3 + tanh ----------------
  v8f acc3[2];
  acc3[0] = {};
  acc3[1] = {};
#pragma unroll
  for (int kt = 0; kt < 4; kt++) {
    const int kk = kt * 32;
    v16h a = ld_a(sFu + rtile * 16 * 128, 128, lane, kk);
#pragma unroll
    for (int n = 0; n < 2; n++) {
      v16h b = ld_b(sWa + (chalf * 2 + n) * 16 * 128, 128, lane, kk);
      acc3[n] = __builtin_amdgcn_wmma_f32_16x16x32_f16(false, a, false, b,
                                                       (short)0, acc3[n], false, false);
    }
  }
  {  // epilogue: + ba1, tanh, store over dead f region
    int rbase = rtile * 16 + ((lane >> 4) ? 8 : 0);
    int cbase = chalf * 32 + (lane & 15);
#pragma unroll
    for (int n = 0; n < 2; n++) {
      int col = cbase + n * 16;
      float bb = ba1[col];
#pragma unroll
      for (int v = 0; v < 8; v++) sA[(rbase + v) * 64 + col] = tanhf(acc3[n][v] + bb);
    }
  }
  __syncthreads();

  // ---------------- Stage 5: logits = a . Wa2 + ba2 ----------------
  {
    int r = tid >> 2, q = tid & 3;
    const float* arow = sA + r * 64 + q * 16;
    float s = 0.f;
#pragma unroll
    for (int i = 0; i < 16; i++) s += arow[i] * Wa2[q * 16 + i];
    s += __shfl_xor(s, 1, 32);
    s += __shfl_xor(s, 2, 32);
    if (q == 0) logits_out[row0 + r] = s + ba2[0];
  }
}

// ---------------- Kernel 2: per-batch masked softmax + top-32 select ----------------
__global__ __launch_bounds__(256)
void select_topk(const float* __restrict__ logits, const unsigned char* __restrict__ mask,
                 const float* __restrict__ fused, float* __restrict__ out_sel,
                 float* __restrict__ out_attn) {
  __shared__ float attn[4096];
  __shared__ float red[256];
  __shared__ int   redi[256];
  __shared__ float sval[32];
  __shared__ int   sidx[32];
  __shared__ float sdenom;
  const int b = blockIdx.x, tid = threadIdx.x;
  const long base = (long)b * 4096;
  const float NEG_INF = -__builtin_inff();

  // masked scaled logits (temperature 0.5 -> *2) + block max
  float lmax = -3.4e38f;
#pragma unroll
  for (int i = 0; i < 16; i++) {
    int t = tid + i * 256;
    float v = mask[base + t] ? logits[base + t] * 2.0f : NEG_INF;
    attn[t] = v;
    lmax = fmaxf(lmax, v);
  }
  red[tid] = lmax;
  __syncthreads();
  for (int s = 128; s > 0; s >>= 1) {
    if (tid < s) red[tid] = fmaxf(red[tid], red[tid + s]);
    __syncthreads();
  }
  float mx = red[0];
  __syncthreads();

  float lsum = 0.f;
#pragma unroll
  for (int i = 0; i < 16; i++) {
    int t = tid + i * 256;
    float e = __expf(attn[t] - mx);  // exp(-inf)=0 for masked
    attn[t] = e;
    lsum += e;
  }
  red[tid] = lsum;
  __syncthreads();
  for (int s = 128; s > 0; s >>= 1) {
    if (tid < s) red[tid] += red[tid + s];
    __syncthreads();
  }
  float inv = 1.0f / red[0];
  __syncthreads();
#pragma unroll
  for (int i = 0; i < 16; i++) {
    int t = tid + i * 256;
    float a = attn[t] * inv;
    attn[t] = a;
    out_attn[base + t] = a;
  }
  __syncthreads();

  // iterative top-32 argmax (attn destroyed after out_attn written)
  for (int k = 0; k < 32; k++) {
    float bv = -1.0f;
    int bi = 0;
#pragma unroll
    for (int i = 0; i < 16; i++) {
      int t = tid + i * 256;
      float v = attn[t];
      if (v > bv) { bv = v; bi = t; }
    }
    red[tid] = bv;
    redi[tid] = bi;
    __syncthreads();
    for (int s = 128; s > 0; s >>= 1) {
      if (tid < s && red[tid + s] > red[tid]) { red[tid] = red[tid + s]; redi[tid] = redi[tid + s]; }
      __syncthreads();
    }
    if (tid == 0) {
      sval[k] = red[0];
      sidx[k] = redi[0];
      attn[redi[0]] = -2.0f;
    }
    __syncthreads();
  }

  // sort index/value pairs ascending by index; denom = sum + 1e-8
  if (tid == 0) {
    for (int i = 1; i < 32; i++) {
      int ki = sidx[i]; float kv = sval[i]; int j = i - 1;
      while (j >= 0 && sidx[j] > ki) { sidx[j + 1] = sidx[j]; sval[j + 1] = sval[j]; j--; }
      sidx[j + 1] = ki; sval[j + 1] = kv;
    }
    float d = 1e-8f;
    for (int i = 0; i < 32; i++) d += sval[i];
    sdenom = d;
  }
  __syncthreads();

  // gather fused rows, scale by renormalized attn
  float invd = 1.0f / sdenom;
#pragma unroll
  for (int i = 0; i < 16; i++) {
    int e = tid * 16 + i;
    int k = e >> 7, dd = e & 127;
    out_sel[((long)b * 32 + k) * 128 + dd] =
        fused[(base + sidx[k]) * 128 + dd] * (sval[k] * invd);
  }
}

// ---------------- Host launcher ----------------
extern "C" void kernel_launch(void* const* d_in, const int* in_sizes, int n_in,
                              void* d_out, int out_size, void* d_ws, size_t ws_size,
                              hipStream_t stream) {
  const float*         X    = (const float*)d_in[0];
  const unsigned char* mask = (const unsigned char*)d_in[1];  // jnp bool -> 1 byte
  const float*         W1   = (const float*)d_in[2];
  const float*         b1   = (const float*)d_in[3];
  const float*         W2   = (const float*)d_in[4];
  const float*         b2   = (const float*)d_in[5];
  const float*         g    = (const float*)d_in[6];
  const float*         be   = (const float*)d_in[7];
  const float*         Wa1  = (const float*)d_in[8];
  const float*         ba1  = (const float*)d_in[9];
  const float*         Wa2  = (const float*)d_in[10];
  const float*         ba2  = (const float*)d_in[11];
  (void)in_sizes; (void)n_in; (void)out_size; (void)ws_size;

  char* ws = (char*)d_ws;
  _Float16* W1T  = (_Float16*)(ws);            // 256*384 f16
  _Float16* W2T  = (_Float16*)(ws + 196608);   // 128*256 f16
  _Float16* Wa1T = (_Float16*)(ws + 262144);   // 64*128 f16
  float* fusedg  = (float*)(ws + 524288);                               // [B*T][128] f32
  float* logitsg = (float*)(ws + 524288 + (size_t)64 * 4096 * 128 * 4); // [B*T] f32

  float* out_sel  = (float*)d_out;             // [64][32][128]
  float* out_attn = out_sel + 64 * 32 * 128;   // [64][4096]

  convert_w<<<544, 256, 0, stream>>>(W1, W2, Wa1, W1T, W2T, Wa1T);
  fuse_kernel<<<4096, 256, 0, stream>>>(X, W1T, W2T, Wa1T, b1, b2, g, be, ba1,
                                        Wa2, ba2, fusedg, logitsg);
  select_topk<<<64, 256, 0, stream>>>(logitsg, mask, fusedg, out_sel, out_attn);
}
